// BiRNN_15616501088715
// MI455X (gfx1250) — matrix operand order repaired and checked by reading, not compiled, not run on hardware
//
#include <hip/hip_runtime.h>
#include <hip/hip_bf16.h>

#define DEVINL __device__ __forceinline__

// ---- plain ext-vector types (trivially constructible, union-safe) ----
typedef __attribute__((ext_vector_type(16))) __bf16   v16bf;
typedef __attribute__((ext_vector_type(8)))  float    v8f;
typedef __attribute__((ext_vector_type(4)))  unsigned u32x4;
typedef __attribute__((ext_vector_type(4)))  int      i32x4;
typedef __attribute__((ext_vector_type(8)))  int      i32x8;

namespace {
constexpr int kB = 64, kT = 512, kE = 256, kH = 256, kO = 256;

// workspace byte offsets
constexpr size_t kOffWxhF = 0;
constexpr size_t kOffWxhB = kOffWxhF + (size_t)kE * kH * 2;
constexpr size_t kOffWhhF = kOffWxhB + (size_t)kE * kH * 2;
constexpr size_t kOffWhhB = kOffWhhF + (size_t)kH * kH * 2;
constexpr size_t kOffWhy  = kOffWhhB + (size_t)kH * kH * 2;
constexpr size_t kOffXfr  = kOffWhy  + (size_t)(2 * kH) * kO * 2;
constexpr size_t kOffXpF  = kOffXfr  + (size_t)kB * kT * kE * 2;
constexpr size_t kOffXpB  = kOffXpF  + (size_t)kT * kB * kH * 4;
constexpr size_t kOffHf   = kOffXpB  + (size_t)kT * kB * kH * 4;
constexpr size_t kOffHb   = kOffHf   + (size_t)kT * kB * kH * 2;
} // namespace

DEVINL __bf16 f2bf(float f) {
  union { float f; unsigned u; } x; x.f = f;
  unsigned r = x.u + 0x7FFFu + ((x.u >> 16) & 1u);   // round-to-nearest-even
  unsigned short h = (unsigned short)(r >> 16);
  return __builtin_bit_cast(__bf16, h);
}

union Frag {            // one lane's WMMA operand: 16 bf16 = 2x b128 loads
  u32x4 u[2];
  v16bf v;
};

DEVINL v8f wmma_bf16(v16bf a, v16bf b, v8f c) {
  // (neg_a, A, neg_b, B, c_mod, C, reuse_a, reuse_b)
  return __builtin_amdgcn_wmma_f32_16x16x32_bf16(false, a, false, b,
                                                 (short)0, c, false, false);
}

// TDM: DMA `bytes` (must be 2*32768*dim1 layout below) from global -> LDS.
// Descriptor per CDNA5 ISA: 2-D tensor of bf16,
// dim0 = 32768 elems, dim1 = nrows, tile == tensor, data_size = 2B, type=2.
DEVINL void tdm_load_to_lds(const void* gsrc, void* lds_dst, unsigned nrows) {
  unsigned lds_off = (unsigned)(size_t)lds_dst;        // low 32 bits = LDS offset
  unsigned long long ga = (unsigned long long)gsrc;
  u32x4 g0;
  g0[0] = 1u;                                          // count=1, no gather
  g0[1] = lds_off;                                     // D#.lds_addr
  g0[2] = (unsigned)(ga & 0xffffffffu);                // global_addr[31:0]
  g0[3] = (unsigned)((ga >> 32) & 0x01ffffffu) | (2u << 30);  // addr hi | type=2
  i32x8 g1;
  g1[0] = (int)(1u << 16);                             // wg_mask=0, data_size=2B
  g1[1] = (int)(32768u << 16);                         // tensor_dim0 lo16 (32768)
  g1[2] = (int)((32768u >> 16) | (nrows << 16));       // dim0 hi16 | dim1 lo16
  g1[3] = (int)(32768u << 16);                         // dim1 hi16 | tile_dim0
  g1[4] = (int)nrows;                                  // tile_dim1 | tile_dim2=0
  g1[5] = 32768;                                       // tensor_dim0_stride lo32
  g1[6] = 0;
  g1[7] = 0;
  i32x4 g2 = {};                                       // unused (<=2D tensor)
  i32x4 g3 = {};
#if defined(__clang_major__) && (__clang_major__ >= 23)
  i32x8 g4 = {};
  __builtin_amdgcn_tensor_load_to_lds(g0, g1, g2, g3, g4, 0);
#else
  __builtin_amdgcn_tensor_load_to_lds(g0, g1, g2, g3, 0);
#endif
}

// ---------------------------------------------------------------------------
// Fragment layouts (consistent producer/consumer conventions):
//  B-matrix (K x N), 32x16 bf16 tile: elem j of lane L = W[kt*32+koff][nt*16+(L%16)], koff as for A below
//  A-matrix (M x K), 16x32 bf16 tile: elem j of lane L = A[mt*16+(L%16)][kt*32+koff],
//     koff = (j<8) ? (L/16)*8+j : 16+(L/16)*8+(j-8)
//  C/D f32 16x16: vgpr r, lane L -> M = r + 8*(L/16), N = L%16
// ---------------------------------------------------------------------------

__global__ void pack_w_frag(const float* __restrict__ W, __bf16* __restrict__ out,
                            int K, int N) {
  int idx = blockIdx.x * blockDim.x + threadIdx.x;
  if (idx >= K * N) return;
  int j  = idx & 15;
  int L  = (idx >> 4) & 31;
  int ntiles = N >> 4;
  int nt = (idx >> 9) % ntiles;
  int kt = idx / (512 * ntiles);
  int k = kt * 32 + ((j < 8) ? ((L >> 4) << 3) + j : 16 + ((L >> 4) << 3) + (j - 8));
  int n = nt * 16 + (L & 15);
  out[idx] = f2bf(W[(size_t)k * N + n]);
}

// X[b,t,e] -> A-frag layout over rows m = t*B + b (tile mt has fixed t)
__global__ void pack_x_frag(const float* __restrict__ X, __bf16* __restrict__ out) {
  int idx = blockIdx.x * blockDim.x + threadIdx.x;   // B*T*E elements
  int j  = idx & 15;
  int L  = (idx >> 4) & 31;
  int kt = (idx >> 9) & 7;          // E/32 = 8 ktiles
  int mt = idx >> 12;
  int m = mt * 16 + (L & 15);
  int t = m >> 6;                   // m / B
  int b = m & 63;                   // m % B
  int half = L >> 4;
  int koff = (j < 8) ? half * 8 + j : 16 + half * 8 + (j - 8);
  out[idx] = f2bf(X[((size_t)b * kT + t) * kE + kt * 32 + koff]);
}

// xproj[dir][t,b,h] = X[b, tsrc, :] @ Wxh + bias ; 256 thr = 8 waves, 64x256 tile
__global__ __launch_bounds__(256) void xproj_kernel(
    const __bf16* __restrict__ Xfrag,
    const __bf16* __restrict__ WfragF, const __bf16* __restrict__ WfragB,
    const float* __restrict__ biasF, const float* __restrict__ biasB,
    float* __restrict__ xpF, float* __restrict__ xpB) {
  int wg  = blockIdx.x;             // 1024 WGs: 512 per direction (one per t)
  int dir = wg >> 9;
  int t   = wg & 511;
  int tsrc = dir ? (kT - 1 - t) : t;
  const __bf16* Wfrag = dir ? WfragB : WfragF;
  const float*  bias  = dir ? biasB  : biasF;
  float*        out   = dir ? xpB    : xpF;

  int lane = threadIdx.x & 31;
  int wv   = threadIdx.x >> 5;      // 8 waves
  int mtile = wv >> 1;              // 4 batch sub-tiles
  int ntb   = (wv & 1) * 8;         // 8 n-tiles per wave

  const u32x4* X4 = (const u32x4*)Xfrag;
  const u32x4* W4 = (const u32x4*)Wfrag;
  int mtG = tsrc * 4 + mtile;

  v8f acc[8] = {};
#pragma unroll 2
  for (int kt = 0; kt < 8; ++kt) {
    Frag a;
    const u32x4* ap = X4 + ((size_t)(mtG * 8 + kt) * 32 + lane) * 2;
    a.u[0] = ap[0]; a.u[1] = ap[1];
#pragma unroll
    for (int q = 0; q < 8; ++q) {
      Frag bf;
      const u32x4* bp = W4 + ((size_t)(kt * 16 + ntb + q) * 32 + lane) * 2;
      bf.u[0] = bp[0]; bf.u[1] = bp[1];
      acc[q] = wmma_bf16(a.v, bf.v, acc[q]);
    }
  }
  int half = lane >> 4, lrow = lane & 15;
#pragma unroll
  for (int q = 0; q < 8; ++q) {
    int n = (ntb + q) * 16 + lrow;
    float bv = bias[n];
#pragma unroll
    for (int r = 0; r < 8; ++r) {
      int brow = mtile * 16 + 8 * half + r;
      out[((size_t)t * kB + brow) * kH + n] = acc[q][r] + bv;
    }
  }
}

// Persistent recurrence: one 32-wave workgroup per direction.
// W_hh fragments DMA'd into LDS once by the Tensor Data Mover (128 KB),
// h (64x256) double-buffered in LDS in A-frag layout (64 KB). 192 KB total.
__global__ __launch_bounds__(1024) void rnn_kernel(
    const __bf16* __restrict__ WhhFragF, const __bf16* __restrict__ WhhFragB,
    const float* __restrict__ xpF, const float* __restrict__ xpB,
    __bf16* __restrict__ hfSeq, __bf16* __restrict__ hbSeq,
    float* __restrict__ lastBase /* d_out + B*T*O */) {
  __shared__ __bf16 sWhh[8 * 16 * 32 * 16];   // 65536 bf16 = 128 KB, frag order
  __shared__ __bf16 sH[2][4 * 8 * 32 * 16];   // 2 x 16384 bf16 = 64 KB

  int dir = blockIdx.x;
  const __bf16* Whh  = dir ? WhhFragB : WhhFragF;
  const float*  xp   = dir ? xpB : xpF;
  __bf16*       hseq = dir ? hbSeq : hfSeq;
  float*        hlast = lastBase + (size_t)dir * kB * kH;

  // Wave 0 kicks the TDM for the whole 128 KB W_hh fragment buffer and waits
  // on TENSORcnt; everyone else zero-fills h0 meanwhile; barrier joins them.
  if (threadIdx.x < 32) {
    tdm_load_to_lds(Whh, sWhh, 2u);           // 2 rows x 32768 bf16 = 128 KB
    __builtin_amdgcn_s_wait_tensorcnt(0);
  } else {
    for (int i = threadIdx.x - 32; i < 4 * 8 * 32 * 16; i += 1024 - 32)
      sH[0][i] = f2bf(0.0f);                  // h0 = 0
  }
  __syncthreads();

  int lane = threadIdx.x & 31;
  int wv   = threadIdx.x >> 5;                // 32 waves, 2 tiles each
  int mtile = wv >> 3;                        // 4 batch sub-tiles
  int nt0   = (wv & 7) * 2;
  int half = lane >> 4, lrow = lane & 15;
  int n0 = nt0 * 16 + lrow;
  int n1 = n0 + 16;

  const u32x4* W4 = (const u32x4*)sWhh;
  int cur = 0;
  for (int t = 0; t < kT; ++t) {
    // Preload this step's xproj values so global latency hides under WMMAs.
    float xv[2][8];
#pragma unroll
    for (int r = 0; r < 8; ++r) {
      int brow = mtile * 16 + 8 * half + r;
      const float* xr = xp + ((size_t)t * kB + brow) * kH;
      xv[0][r] = xr[n0];
      xv[1][r] = xr[n1];
      __builtin_prefetch(xr + kB * kH + n0, 0, 0);   // next timestep
    }

    v8f c0 = {}, c1 = {};
    const u32x4* H4 = (const u32x4*)sH[cur];
#pragma unroll
    for (int kt = 0; kt < 8; ++kt) {
      Frag a;
      const u32x4* ap = H4 + ((mtile * 8 + kt) * 32 + lane) * 2;
      a.u[0] = ap[0]; a.u[1] = ap[1];
      Frag b0, b1;
      const u32x4* bp0 = W4 + ((kt * 16 + nt0) * 32 + lane) * 2;
      const u32x4* bp1 = W4 + ((kt * 16 + nt0 + 1) * 32 + lane) * 2;
      b0.u[0] = bp0[0]; b0.u[1] = bp0[1];
      b1.u[0] = bp1[0]; b1.u[1] = bp1[1];
      c0 = wmma_bf16(a.v, b0.v, c0);
      c1 = wmma_bf16(a.v, b1.v, c1);
    }
    int tStore = dir ? (kT - 1 - t) : t;      // reference reverses hb in time
#pragma unroll
    for (int s = 0; s < 2; ++s) {
      v8f& c = s ? c1 : c0;
      int n = s ? n1 : n0;
      int kt2 = n >> 5, klocal = n & 31;
      // target (lane,elem) inside next-step A-frag for column k = n
      int j2    = (klocal & 7) + ((klocal & 16) ? 8 : 0);
      int L2add = (klocal & 8) ? 16 : 0;
#pragma unroll
      for (int r = 0; r < 8; ++r) {
        int brow = mtile * 16 + 8 * half + r;
        float hv = tanhf(xv[s][r] + c[r]);
        __bf16 h16 = f2bf(hv);
        int rowL = brow & 15;
        sH[cur ^ 1][((mtile * 8 + kt2) * 32 + (rowL + L2add)) * 16 + j2] = h16;
        hseq[((size_t)tStore * kB + brow) * kH + n] = h16;
        if (t == kT - 1) hlast[(size_t)brow * kH + n] = hv;  // hf_last / hb_last
      }
    }
    __syncthreads();
    cur ^= 1;
  }
}

// Y[b,t,o] = concat(hf,hb)[t,b,:] @ W_hy + b_y ; rows m = b*T + t
__global__ __launch_bounds__(256) void out_kernel(
    const __bf16* __restrict__ hf, const __bf16* __restrict__ hb,
    const __bf16* __restrict__ WhyFrag, const float* __restrict__ by,
    float* __restrict__ Y) {
  int g = blockIdx.x;                        // 512 WGs, 64 rows x 256 cols each
  int lane = threadIdx.x & 31;
  int wv   = threadIdx.x >> 5;
  int mt   = g * 4 + (wv >> 1);
  int ntb  = (wv & 1) * 8;
  int half = lane >> 4, lrow = lane & 15;

  int m = mt * 16 + lrow;                    // this lane's A-row
  int b = m >> 9;                            // m / T
  int t = m & 511;                           // m % T
  size_t rowOff = ((size_t)t * kB + b) * kH; // into [t][b][h] bf16 arrays

  const u32x4* W4 = (const u32x4*)WhyFrag;
  v8f acc[8] = {};
#pragma unroll 2
  for (int kt = 0; kt < 16; ++kt) {          // K = 2H = 512
    int k0 = kt * 32 + half * 8;
    int k1 = k0 + 16;
    const __bf16* s0 = (k0 < kH) ? hf + rowOff + k0 : hb + rowOff + (k0 - kH);
    const __bf16* s1 = (k1 < kH) ? hf + rowOff + k1 : hb + rowOff + (k1 - kH);
    Frag a;
    a.u[0] = *(const u32x4*)s0;
    a.u[1] = *(const u32x4*)s1;
#pragma unroll
    for (int q = 0; q < 8; ++q) {
      Frag bf;
      const u32x4* bp = W4 + ((size_t)(kt * 16 + ntb + q) * 32 + lane) * 2;
      bf.u[0] = bp[0]; bf.u[1] = bp[1];
      acc[q] = wmma_bf16(a.v, bf.v, acc[q]);
    }
  }
#pragma unroll
  for (int q = 0; q < 8; ++q) {
    int n = (ntb + q) * 16 + lrow;
    float bv = by[n];
#pragma unroll
    for (int r = 0; r < 8; ++r) {
      int mrow = mt * 16 + 8 * half + r;
      Y[(size_t)mrow * kO + n] = acc[q][r] + bv;
    }
  }
}

extern "C" void kernel_launch(void* const* d_in, const int* in_sizes, int n_in,
                              void* d_out, int out_size, void* d_ws, size_t ws_size,
                              hipStream_t stream) {
  (void)in_sizes; (void)n_in; (void)out_size; (void)ws_size;
  const float* X    = (const float*)d_in[0];
  const float* WxhF = (const float*)d_in[1];
  const float* WhhF = (const float*)d_in[2];
  const float* bF   = (const float*)d_in[3];
  const float* WxhB = (const float*)d_in[4];
  const float* WhhB = (const float*)d_in[5];
  const float* bB   = (const float*)d_in[6];
  const float* Why  = (const float*)d_in[7];
  const float* bY   = (const float*)d_in[8];
  float* Y = (float*)d_out;

  char* ws = (char*)d_ws;
  __bf16* wxhF_p = (__bf16*)(ws + kOffWxhF);
  __bf16* wxhB_p = (__bf16*)(ws + kOffWxhB);
  __bf16* whhF_p = (__bf16*)(ws + kOffWhhF);
  __bf16* whhB_p = (__bf16*)(ws + kOffWhhB);
  __bf16* why_p  = (__bf16*)(ws + kOffWhy);
  __bf16* xfr_p  = (__bf16*)(ws + kOffXfr);
  float*  xpF_p  = (float*)(ws + kOffXpF);
  float*  xpB_p  = (float*)(ws + kOffXpB);
  __bf16* hf_p   = (__bf16*)(ws + kOffHf);
  __bf16* hb_p   = (__bf16*)(ws + kOffHb);

  // 1) pack weights + X into bf16 WMMA fragment layouts
  pack_w_frag<<<(kE * kH + 255) / 256, 256, 0, stream>>>(WxhF, wxhF_p, kE, kH);
  pack_w_frag<<<(kE * kH + 255) / 256, 256, 0, stream>>>(WxhB, wxhB_p, kE, kH);
  pack_w_frag<<<(kH * kH + 255) / 256, 256, 0, stream>>>(WhhF, whhF_p, kH, kH);
  pack_w_frag<<<(kH * kH + 255) / 256, 256, 0, stream>>>(WhhB, whhB_p, kH, kH);
  pack_w_frag<<<(2 * kH * kO + 255) / 256, 256, 0, stream>>>(Why, why_p, 2 * kH, kO);
  pack_x_frag<<<(kB * kT * kE) / 256, 256, 0, stream>>>(X, xfr_p);

  // 2) both input projections (parallel GEMM, WMMA bf16)
  xproj_kernel<<<1024, 256, 0, stream>>>(xfr_p, wxhF_p, wxhB_p, bF, bB, xpF_p, xpB_p);

  // 3) the sequential recurrences: one persistent 32-wave WG per direction
  rnn_kernel<<<2, 1024, 0, stream>>>(whhF_p, whhB_p, xpF_p, xpB_p, hf_p, hb_p,
                                     Y + (size_t)kB * kT * kO);

  // 4) output projection (parallel GEMM, WMMA bf16)
  out_kernel<<<512, 256, 0, stream>>>(hf_p, hb_p, why_p, bY, Y);
}
